// MultiHeadCrossAttention_11922829214447
// MI455X (gfx1250) — matrix operand order repaired and checked by reading, not compiled, not run on hardware
//
#include <hip/hip_runtime.h>
#include <hip/hip_bf16.h>

// Problem constants (match reference)
#define D_MODEL   1024
#define NUM_HEADS 16
#define DEPTH     64
#define BATCH     4
#define L_Q       1024
#define L_KV      2048

typedef __attribute__((ext_vector_type(8)))  float  v8f;
typedef __attribute__((ext_vector_type(8)))  __bf16 v8bf;
typedef __attribute__((ext_vector_type(16))) __bf16 v16bf;

// ---------------------------------------------------------------------------
// fp32 -> bf16, round-to-nearest-even
// ---------------------------------------------------------------------------
__device__ __forceinline__ __bf16 f2bf(float f) {
  unsigned u;
  __builtin_memcpy(&u, &f, 4);
  u += 0x7fffu + ((u >> 16) & 1u);
  unsigned short h = (unsigned short)(u >> 16);
  __bf16 b;
  __builtin_memcpy(&b, &h, 2);
  return b;
}

__global__ void pack_bf16_kernel(const float* __restrict__ in,
                                 __bf16* __restrict__ out, int n) {
  int i = blockIdx.x * blockDim.x + threadIdx.x;
  int stride = gridDim.x * blockDim.x;
  for (; i < n; i += stride) out[i] = f2bf(in[i]);
}

// ---------------------------------------------------------------------------
// WMMA fragment loaders (wave32, 16x16x32 bf16)
//
// A operand (16x32, M x K): lane&15 = row M.
//   regs 0-3: k = 8*laneHi + {0..7}; regs 4-7: k = 16 + 8*laneHi + {0..7}
//   -> two contiguous 16B chunks at k0 + 8*lh and k0 + 16 + 8*lh.
// B operand (32x16, K x N): lane&15 = column N.
//   same k order as A: chunks at k0 + 8*lh and k0 + 16 + 8*lh.
// ---------------------------------------------------------------------------
__device__ __forceinline__ v16bf load_frag_a(const __bf16* base, int stride,
                                             int m0, int k0) {
  const int lane = threadIdx.x & 31;
  const int ln = lane & 15, lh = lane >> 4;
  const __bf16* p = base + (size_t)(m0 + ln) * stride + k0 + lh * 8;
  v8bf lo = *(const v8bf*)p;
  v8bf hi = *(const v8bf*)(p + 16);
  return __builtin_shufflevector(lo, hi, 0, 1, 2, 3, 4, 5, 6, 7,
                                 8, 9, 10, 11, 12, 13, 14, 15);
}

__device__ __forceinline__ v16bf load_frag_b(const __bf16* base, int stride,
                                             int n0, int k0) {
  const int lane = threadIdx.x & 31;
  const int ln = lane & 15, lh = lane >> 4;
  const __bf16* p = base + (size_t)(n0 + ln) * stride + k0 + lh * 8;
  v8bf lo = *(const v8bf*)p;
  v8bf hi = *(const v8bf*)(p + 16);
  return __builtin_shufflevector(lo, hi, 0, 1, 2, 3, 4, 5, 6, 7,
                                 8, 9, 10, 11, 12, 13, 14, 15);
}

__device__ __forceinline__ v8f wmma_bf16(v16bf a, v16bf b, v8f c) {
  return __builtin_amdgcn_wmma_f32_16x16x32_bf16(false, a, false, b,
                                                 (short)0, c, false, false);
}

// ---------------------------------------------------------------------------
// C/D tile store. C layout: lane&15 = column N, reg r = row (r + 8*laneHi).
// MODE 0: fp32 row-major [M,N], out = scale*(acc+bias)  — dense projection
// MODE 1: bf16 row-major [M,N]                          — q/k projections
// MODE 2: bf16 transposed per batch -> vt[(b*N + n)*L_KV + m%L_KV]
//         (8 consecutive rows per lane are contiguous -> one 16B store)
// ---------------------------------------------------------------------------
template <int MODE>
__device__ __forceinline__ void store_tile(v8f acc, void* out, int N,
                                           int m0, int n0,
                                           const float* __restrict__ bias,
                                           float oscale) {
  const int lane = threadIdx.x & 31;
  const int ln = lane & 15, lh = lane >> 4;
  const int n = n0 + ln;
  const float bv = bias ? bias[n] : 0.0f;
  const int mrow = m0 + lh * 8;
  if constexpr (MODE == 0) {
    float* o = (float*)out;
#pragma unroll
    for (int r = 0; r < 8; ++r)
      o[(size_t)(mrow + r) * N + n] = (acc[r] + bv) * oscale;
  } else if constexpr (MODE == 1) {
    __bf16* o = (__bf16*)out;
#pragma unroll
    for (int r = 0; r < 8; ++r)
      o[(size_t)(mrow + r) * N + n] = f2bf((acc[r] + bv) * oscale);
  } else {
    __bf16* o = (__bf16*)out;
    const int bb = mrow / L_KV;
    const int l = mrow - bb * L_KV;
    v8bf pk;
#pragma unroll
    for (int r = 0; r < 8; ++r) pk[r] = f2bf((acc[r] + bv) * oscale);
    *(v8bf*)(o + ((size_t)bb * N + n) * L_KV + l) = pk;
  }
}

// ---------------------------------------------------------------------------
// GEMM: C[M,N] = scale*(A[M,K] @ W[N,K]^T + bias).
// Block = 4 waves = 128x128 tile; each wave owns 64x64 (4x4 WMMA tiles).
// Fragments are double-buffered: 8 loads per k-step feed 16 WMMAs.
// ---------------------------------------------------------------------------
template <int MODE>
__global__ void __launch_bounds__(128)
gemm_xWT(const __bf16* __restrict__ A, const __bf16* __restrict__ W,
         const float* __restrict__ bias, void* __restrict__ out,
         int N, int K, float oscale) {
  const int wave = threadIdx.x >> 5;
  const int row0 = blockIdx.y * 128 + (wave >> 1) * 64;
  const int col0 = blockIdx.x * 128 + (wave & 1) * 64;
  v8f acc[4][4];
#pragma unroll
  for (int i = 0; i < 4; ++i)
#pragma unroll
    for (int j = 0; j < 4; ++j) acc[i][j] = (v8f){};

  v16bf fa[4], fb[4];
#pragma unroll
  for (int i = 0; i < 4; ++i) {
    fa[i] = load_frag_a(A, K, row0 + 16 * i, 0);
    fb[i] = load_frag_b(W, K, col0 + 16 * i, 0);
  }

  for (int k0 = 0; k0 < K; k0 += 32) {
    const int kn = (k0 + 32 < K) ? (k0 + 32) : k0;
    v16bf na[4], nb[4];
#pragma unroll
    for (int i = 0; i < 4; ++i) {
      na[i] = load_frag_a(A, K, row0 + 16 * i, kn);
      nb[i] = load_frag_b(W, K, col0 + 16 * i, kn);
    }
#pragma unroll
    for (int i = 0; i < 4; ++i)
#pragma unroll
      for (int j = 0; j < 4; ++j)
        acc[i][j] = wmma_bf16(fa[i], fb[j], acc[i][j]);
#pragma unroll
    for (int i = 0; i < 4; ++i) {
      fa[i] = na[i];
      fb[i] = nb[i];
    }
  }
#pragma unroll
  for (int i = 0; i < 4; ++i)
#pragma unroll
    for (int j = 0; j < 4; ++j)
      store_tile<MODE>(acc[i][j], out, N, row0 + 16 * i, col0 + 16 * j, bias,
                       oscale);
}

// ---------------------------------------------------------------------------
// Fused flash attention. One wave owns 16 query rows of one (b,h).
// q_p arrives pre-scaled by 1/sqrt(DEPTH). Per 32-kv tile:
//   - V fragments issued FIRST (overlap softmax),
//   - S = Q K^T via 4 WMMAs on double-buffered K fragments,
//   - online softmax: each lane owns half a row (16 cols); partial max/sum
//     exchanged through LDS (same-wave ordering -> s_wait_dscnt only),
//   - next tile's K fragments issued before the PV WMMAs (overlap),
//   - P re-read from LDS in A-operand layout, PV via 4 WMMAs into fp32 O.
// ---------------------------------------------------------------------------
__global__ void __launch_bounds__(128)
attn_kernel(const __bf16* __restrict__ qp, const __bf16* __restrict__ kp,
            const __bf16* __restrict__ vt, __bf16* __restrict__ ctx) {
  __shared__ __align__(32) float  sS[4][16 * 32];  // scores
  __shared__ __align__(32) __bf16 sP[4][16 * 32];  // exp(S - m) in bf16
  __shared__ __align__(32) float  sMx[4][32];      // partial row max
  __shared__ __align__(32) float  sSm[4][32];      // partial row sum
  __shared__ __align__(32) float  sR[4][16];       // per-row broadcast

  const int wave = threadIdx.x >> 5;
  const int lane = threadIdx.x & 31;
  const int ln = lane & 15, lh = lane >> 4;
  const int b = blockIdx.z, h = blockIdx.y;
  const int q0 = blockIdx.x * 64 + wave * 16;
  const int qrow = b * L_Q + q0;
  const int doff = h * DEPTH;
  const int vrow = b * D_MODEL + doff;

  float* Sw = sS[wave];
  __bf16* Pw = sP[wave];
  float* Mw = sMx[wave];
  float* Uw = sSm[wave];
  float* Rw = sR[wave];

  // Q fragments for the whole kv sweep (16 rows x 64 depth), pre-scaled.
  const v16bf aq0 = load_frag_a(qp, D_MODEL, qrow, doff);
  const v16bf aq1 = load_frag_a(qp, D_MODEL, qrow, doff + 32);

  // Preload K fragments for the first kv tile.
  v16bf bk00 = load_frag_b(kp, D_MODEL, b * L_KV, doff);
  v16bf bk01 = load_frag_b(kp, D_MODEL, b * L_KV, doff + 32);
  v16bf bk10 = load_frag_b(kp, D_MODEL, b * L_KV + 16, doff);
  v16bf bk11 = load_frag_b(kp, D_MODEL, b * L_KV + 16, doff + 32);

  v8f o0 = {}, o1 = {}, o2 = {}, o3 = {};
  float m_i = -1e30f, l_i = 0.0f;

  for (int kv0 = 0; kv0 < L_KV; kv0 += 32) {
    // Issue V fragment loads early; consumed only after softmax.
    v16bf bv0 = load_frag_b(vt, L_KV, vrow, kv0);
    v16bf bv1 = load_frag_b(vt, L_KV, vrow + 16, kv0);
    v16bf bv2 = load_frag_b(vt, L_KV, vrow + 32, kv0);
    v16bf bv3 = load_frag_b(vt, L_KV, vrow + 48, kv0);

    v8f s0 = {}, s1 = {};
    s0 = wmma_bf16(aq0, bk00, s0);
    s0 = wmma_bf16(aq1, bk01, s0);
    s1 = wmma_bf16(aq0, bk10, s1);
    s1 = wmma_bf16(aq1, bk11, s1);

    // Spill S to LDS in plain [row][kv] layout.
#pragma unroll
    for (int r = 0; r < 8; ++r) {
      Sw[(r + 8 * lh) * 32 + ln] = s0[r];
      Sw[(r + 8 * lh) * 32 + 16 + ln] = s1[r];
    }
    asm volatile("s_wait_dscnt 0" ::: "memory");

    // Online softmax: lane owns row ln, column half lh (16 columns).
    const float* srow = &Sw[ln * 32 + lh * 16];
    float pmax = -1e30f;
#pragma unroll
    for (int t = 0; t < 16; ++t) pmax = fmaxf(pmax, srow[t]);
    Mw[lh * 16 + ln] = pmax;
    asm volatile("s_wait_dscnt 0" ::: "memory");
    const float rowmax = fmaxf(pmax, Mw[(1 - lh) * 16 + ln]);
    const float mnew = fmaxf(m_i, rowmax);
    const float corr = __expf(m_i - mnew);
    float psum = 0.0f;
    v8bf e0, e1;
#pragma unroll
    for (int t = 0; t < 8; ++t) {
      const float e = __expf(srow[t] - mnew);
      psum += e;
      e0[t] = f2bf(e);
    }
#pragma unroll
    for (int t = 0; t < 8; ++t) {
      const float e = __expf(srow[8 + t] - mnew);
      psum += e;
      e1[t] = f2bf(e);
    }
    __bf16* prow = &Pw[ln * 32 + lh * 16];
    *(v8bf*)prow = e0;
    *(v8bf*)(prow + 8) = e1;
    Uw[lh * 16 + ln] = psum;
    asm volatile("s_wait_dscnt 0" ::: "memory");
    l_i = l_i * corr + psum + Uw[(1 - lh) * 16 + ln];
    m_i = mnew;
    if (lh == 0) Rw[ln] = corr;
    asm volatile("s_wait_dscnt 0" ::: "memory");

    // Prefetch next tile's K fragments; overlap the PV WMMAs below.
    const int kvn = (kv0 + 32 < L_KV) ? (kv0 + 32) : kv0;
    const int krown = b * L_KV + kvn;
    v16bf nk00 = load_frag_b(kp, D_MODEL, krown, doff);
    v16bf nk01 = load_frag_b(kp, D_MODEL, krown, doff + 32);
    v16bf nk10 = load_frag_b(kp, D_MODEL, krown + 16, doff);
    v16bf nk11 = load_frag_b(kp, D_MODEL, krown + 16, doff + 32);

    // Rescale O accumulators by the per-row correction factor.
    float cr[8];
#pragma unroll
    for (int r = 0; r < 8; ++r) cr[r] = Rw[r + 8 * lh];
#pragma unroll
    for (int r = 0; r < 8; ++r) {
      o0[r] *= cr[r]; o1[r] *= cr[r]; o2[r] *= cr[r]; o3[r] *= cr[r];
    }

    // P (16x32) as A operand straight from LDS; PV accumulation.
    const v16bf pa = load_frag_a(Pw, 32, 0, 0);
    o0 = wmma_bf16(pa, bv0, o0);
    o1 = wmma_bf16(pa, bv1, o1);
    o2 = wmma_bf16(pa, bv2, o2);
    o3 = wmma_bf16(pa, bv3, o3);

    bk00 = nk00; bk01 = nk01; bk10 = nk10; bk11 = nk11;
  }

  // Final normalization and store to ctx[b*L_Q + q, h*64 + d] (bf16).
  if (lh == 0) Rw[ln] = 1.0f / l_i;
  asm volatile("s_wait_dscnt 0" ::: "memory");
  float inv[8];
#pragma unroll
  for (int r = 0; r < 8; ++r) inv[r] = Rw[r + 8 * lh];
#pragma unroll
  for (int r = 0; r < 8; ++r) {
    const size_t mrow = (size_t)(qrow + 8 * lh + r) * D_MODEL;
    ctx[mrow + doff + ln]      = f2bf(o0[r] * inv[r]);
    ctx[mrow + doff + 16 + ln] = f2bf(o1[r] * inv[r]);
    ctx[mrow + doff + 32 + ln] = f2bf(o2[r] * inv[r]);
    ctx[mrow + doff + 48 + ln] = f2bf(o3[r] * inv[r]);
  }
}

// ---------------------------------------------------------------------------
// Host launcher
// ---------------------------------------------------------------------------
extern "C" void kernel_launch(void* const* d_in, const int* in_sizes, int n_in,
                              void* d_out, int out_size, void* d_ws,
                              size_t ws_size, hipStream_t stream) {
  const float* query   = (const float*)d_in[0];
  const float* value   = (const float*)d_in[1];
  const float* wq_w    = (const float*)d_in[2];
  const float* wk_w    = (const float*)d_in[3];
  const float* wv_w    = (const float*)d_in[4];
  const float* dense_w = (const float*)d_in[5];
  const float* wq_b    = (const float*)d_in[6];
  const float* wk_b    = (const float*)d_in[7];
  const float* wv_b    = (const float*)d_in[8];
  const float* dense_b = (const float*)d_in[9];

  const size_t NQ = (size_t)BATCH * L_Q * D_MODEL;   // 4M elems
  const size_t NV = (size_t)BATCH * L_KV * D_MODEL;  // 8M elems
  const size_t NW = (size_t)D_MODEL * D_MODEL;       // 1M elems

  char* ws = (char*)d_ws;
  size_t off = 0;
  __bf16* xq  = (__bf16*)(ws + off); off += NQ * 2;  // packed query; later ctx
  __bf16* xv  = (__bf16*)(ws + off); off += NV * 2;  // packed value
  __bf16* wqb = (__bf16*)(ws + off); off += NW * 2;
  __bf16* wkb = (__bf16*)(ws + off); off += NW * 2;
  __bf16* wvb = (__bf16*)(ws + off); off += NW * 2;
  __bf16* wdb = (__bf16*)(ws + off); off += NW * 2;
  __bf16* q_p = (__bf16*)(ws + off); off += NQ * 2;  // q projection (pre-scaled)
  __bf16* k_p = (__bf16*)(ws + off); off += NV * 2;  // k projection
  __bf16* vt  = (__bf16*)(ws + off); off += NV * 2;  // v projection, transposed
  __bf16* ctx = xq;  // query dead after q projection; reuse for context

  // 1) fp32 -> bf16 packing
  pack_bf16_kernel<<<2048, 256, 0, stream>>>(query, xq, (int)NQ);
  pack_bf16_kernel<<<2048, 256, 0, stream>>>(value, xv, (int)NV);
  pack_bf16_kernel<<<1024, 256, 0, stream>>>(wq_w, wqb, (int)NW);
  pack_bf16_kernel<<<1024, 256, 0, stream>>>(wk_w, wkb, (int)NW);
  pack_bf16_kernel<<<1024, 256, 0, stream>>>(wv_w, wvb, (int)NW);
  pack_bf16_kernel<<<1024, 256, 0, stream>>>(dense_w, wdb, (int)NW);

  // 2) projections (WMMA GEMMs); q pre-scaled by 1/sqrt(64); v head-transposed
  gemm_xWT<1><<<dim3(D_MODEL / 128, BATCH * L_Q / 128), 128, 0, stream>>>(
      xq, wqb, wq_b, q_p, D_MODEL, D_MODEL, 0.125f);
  gemm_xWT<1><<<dim3(D_MODEL / 128, BATCH * L_KV / 128), 128, 0, stream>>>(
      xv, wkb, wk_b, k_p, D_MODEL, D_MODEL, 1.0f);
  gemm_xWT<2><<<dim3(D_MODEL / 128, BATCH * L_KV / 128), 128, 0, stream>>>(
      xv, wvb, wv_b, vt, D_MODEL, D_MODEL, 1.0f);

  // 3) fused flash attention
  attn_kernel<<<dim3(L_Q / 64, NUM_HEADS, BATCH), 128, 0, stream>>>(
      q_p, k_p, vt, ctx);

  // 4) dense output projection, fp32 out
  gemm_xWT<0><<<dim3(D_MODEL / 128, BATCH * L_Q / 128), 128, 0, stream>>>(
      ctx, wdb, dense_b, d_out, D_MODEL, D_MODEL, 1.0f);

  (void)in_sizes; (void)n_in; (void)out_size; (void)ws_size;
}
